// NoCrossVanillaGATv2EncoderModel_44504451121592
// MI455X (gfx1250) — compile-verified, matrix operand order checked by reading, not run on hardware
//
#include <hip/hip_runtime.h>
#include <hip/hip_bf16.h>

typedef __attribute__((ext_vector_type(16))) _Float16 v16h;
typedef __attribute__((ext_vector_type(8)))  _Float16 v8h;
typedef __attribute__((ext_vector_type(8)))  float    v8f;

#define NEG_SLOPE 0.2f

__device__ __forceinline__ float lrelu(float x) { return x >= 0.f ? x : NEG_SLOPE * x; }

// float atomic max via integer punning (works for mixed signs)
__device__ __forceinline__ void atomicMaxF(float* addr, float v) {
  if (v >= 0.f) atomicMax((int*)addr, __float_as_int(v));
  else          atomicMin((unsigned int*)addr, __float_as_uint(v));
}

// ---------------- utility kernels ----------------

__global__ void k_fill(float* __restrict__ p, int n, float v) {
  int i = blockIdx.x * blockDim.x + threadIdx.x;
  if (i < n) p[i] = v;
}

__global__ void k_f32_to_f16(const float* __restrict__ x, _Float16* __restrict__ y, int n) {
  int i = blockIdx.x * blockDim.x + threadIdx.x;
  if (i < n) y[i] = (_Float16)x[i];
}

// Pack W[K][M] (row-major f32) into WMMA-B per-lane layout:
// pW[((tn*ksteps + ks)*32 + lane)*16 + j]  holds the 16 halfs lane needs for
// v_wmma_f32_16x16x32_f16 B operand (col = tn*16 + lane&15,
// K = ks*32 + (lane>=16 ? 8 : 0) + {j<8 ? j : j+8}).
__global__ void k_pack_w(const float* __restrict__ W, _Float16* __restrict__ pW,
                         int K, int M) {
  int idx = blockIdx.x * blockDim.x + threadIdx.x;
  int total = (M / 16) * (K / 32) * 32;
  if (idx >= total) return;
  int l = idx & 31;
  int g = idx >> 5;
  int ksteps = K / 32;
  int ks = g % ksteps;
  int tn = g / ksteps;
  int n  = tn * 16 + (l & 15);
  int kb = ks * 32 + ((l >> 4) & 1) * 8;
  _Float16* outp = pW + (size_t)idx * 16;
#pragma unroll
  for (int j = 0; j < 16; ++j) {
    int k = kb + (j < 8 ? j : j + 8);   // j>=8 -> kb + 16 + (j-8)
    outp[j] = (_Float16)W[(size_t)k * M + n];
  }
}

// ---------------- WMMA GEMM: C[Nrows x M] = A[Nrows x K](f16) * W + bias ----------------
// grid.x = ceil(Nrows/16), grid.y = M/128, block = 256 (8 waves, one 16x16 tile each)
__global__ void k_gemm_wmma(const _Float16* __restrict__ A,
                            const _Float16* __restrict__ pW,
                            const float* __restrict__ bias,
                            float* __restrict__ C,
                            int Nrows, int K, int M) {
  int wv = threadIdx.x >> 5;
  int l  = threadIdx.x & 31;
  int tm = blockIdx.x;
  int tn = blockIdx.y * 8 + wv;

  int row = tm * 16 + (l & 15);
  if (row >= Nrows) row = Nrows - 1;      // clamp loads; EXEC stays all-1 for WMMA
  int kshift = ((l >> 4) & 1) * 8;
  int ksteps = K / 32;

  v8f c = {};
  const _Float16* pB = pW + ((size_t)tn * ksteps) * 32 * 16 + (size_t)l * 16;
#pragma unroll 4
  for (int ks = 0; ks < ksteps; ++ks) {
    const _Float16* ap = A + (size_t)row * K + ks * 32 + kshift;
    v8h alo = *(const v8h*)ap;
    v8h ahi = *(const v8h*)(ap + 16);
    v16h a = __builtin_shufflevector(alo, ahi,
                                     0, 1, 2, 3, 4, 5, 6, 7,
                                     8, 9, 10, 11, 12, 13, 14, 15);
    v16h b = *(const v16h*)(pB + (size_t)ks * 32 * 16);
    c = __builtin_amdgcn_wmma_f32_16x16x32_f16(false, a, false, b,
                                               (short)0, c, false, false);
  }

  int col = tn * 16 + (l & 15);
  int m0  = tm * 16 + ((l >> 4) & 1) * 8;
  float bv = bias[col];
#pragma unroll
  for (int r = 0; r < 8; ++r) {
    int m = m0 + r;
    if (m < Nrows) C[(size_t)m * M + col] = c[r] + bv;
  }
}

// ---------------- conv1 edge kernels (H=8, C=16, D=128) ----------------

__global__ void k_att1_logits(const float* __restrict__ xl, const float* __restrict__ xr,
                              const float* __restrict__ att,
                              const int* __restrict__ src, const int* __restrict__ dst,
                              float* __restrict__ elog, float* __restrict__ emax,
                              int E, int Etot) {
  int idx = blockIdx.x * blockDim.x + threadIdx.x;
  if (idx >= Etot * 8) return;
  int e = idx >> 3, h = idx & 7;
  int s, d;
  if (e < E) { s = src[e]; d = dst[e]; } else { s = d = e - E; }
  const float4* pl = (const float4*)(xl + (size_t)s * 128 + h * 16);
  const float4* pr = (const float4*)(xr + (size_t)d * 128 + h * 16);
  const float4* pa = (const float4*)(att + h * 16);
  float acc = 0.f;
#pragma unroll
  for (int j = 0; j < 4; ++j) {
    float4 a = pl[j], b = pr[j], w = pa[j];
    acc += w.x * lrelu(a.x + b.x) + w.y * lrelu(a.y + b.y) +
           w.z * lrelu(a.z + b.z) + w.w * lrelu(a.w + b.w);
  }
  elog[idx] = acc;
  atomicMaxF(&emax[d * 8 + h], acc);
}

__global__ void k_att1_exp(float* __restrict__ elog, const float* __restrict__ emax,
                           float* __restrict__ den, const int* __restrict__ dst,
                           int E, int Etot) {
  int idx = blockIdx.x * blockDim.x + threadIdx.x;
  if (idx >= Etot * 8) return;
  int e = idx >> 3, h = idx & 7;
  int d = (e < E) ? dst[e] : (e - E);
  float ex = __expf(elog[idx] - emax[d * 8 + h]);
  elog[idx] = ex;
  atomicAdd(&den[d * 8 + h], ex);
}

__global__ void k_att1_scatter(const float* __restrict__ elog, const float* __restrict__ den,
                               const float* __restrict__ xl,
                               const int* __restrict__ src, const int* __restrict__ dst,
                               float* __restrict__ acc1, int E, int Etot) {
  int idx = blockIdx.x * blockDim.x + threadIdx.x;
  if (idx >= Etot * 8) return;
  int e = idx >> 3, h = idx & 7;
  int s, d;
  if (e < E) { s = src[e]; d = dst[e]; } else { s = d = e - E; }
  float alpha = elog[idx] / den[d * 8 + h];
  const float4* pl = (const float4*)(xl + (size_t)s * 128 + h * 16);
  float* po = acc1 + (size_t)d * 128 + h * 16;
#pragma unroll
  for (int j = 0; j < 4; ++j) {
    float4 a = pl[j];
    atomicAdd(po + 4 * j + 0, alpha * a.x);
    atomicAdd(po + 4 * j + 1, alpha * a.y);
    atomicAdd(po + 4 * j + 2, alpha * a.z);
    atomicAdd(po + 4 * j + 3, alpha * a.w);
  }
}

// bias + relu, emit f16 activations for the next GEMM
__global__ void k_fin1(const float* __restrict__ acc1, const float* __restrict__ bias,
                       _Float16* __restrict__ h1h, int total) {
  int i = blockIdx.x * blockDim.x + threadIdx.x;
  if (i >= total) return;
  float v = acc1[i] + bias[i & 127];
  v = v > 0.f ? v : 0.f;
  h1h[i] = (_Float16)v;
}

// ---------------- conv2 edge kernels (H=1, C=256) ----------------

__global__ void k_att2_logits(const float* __restrict__ xl, const float* __restrict__ xr,
                              const float* __restrict__ att,
                              const int* __restrict__ src, const int* __restrict__ dst,
                              float* __restrict__ elog, float* __restrict__ emax,
                              int E, int Etot) {
  int wave = blockIdx.x * (blockDim.x >> 5) + (threadIdx.x >> 5);
  int l = threadIdx.x & 31;
  if (wave >= Etot) return;
  int s, d;
  if (wave < E) { s = src[wave]; d = dst[wave]; } else { s = d = wave - E; }
  const float4* pl = (const float4*)(xl + (size_t)s * 256 + l * 8);
  const float4* pr = (const float4*)(xr + (size_t)d * 256 + l * 8);
  const float4* pa = (const float4*)(att + l * 8);
  float acc = 0.f;
#pragma unroll
  for (int j = 0; j < 2; ++j) {
    float4 a = pl[j], b = pr[j], w = pa[j];
    acc += w.x * lrelu(a.x + b.x) + w.y * lrelu(a.y + b.y) +
           w.z * lrelu(a.z + b.z) + w.w * lrelu(a.w + b.w);
  }
#pragma unroll
  for (int o = 16; o >= 1; o >>= 1) acc += __shfl_xor(acc, o, 32);
  if (l == 0) {
    elog[wave] = acc;
    atomicMaxF(&emax[d], acc);
  }
}

__global__ void k_att2_exp(float* __restrict__ elog, const float* __restrict__ emax,
                           float* __restrict__ den, const int* __restrict__ dst,
                           int E, int Etot) {
  int e = blockIdx.x * blockDim.x + threadIdx.x;
  if (e >= Etot) return;
  int d = (e < E) ? dst[e] : (e - E);
  float ex = __expf(elog[e] - emax[d]);
  elog[e] = ex;
  atomicAdd(&den[d], ex);
}

__global__ void k_att2_scatter(const float* __restrict__ elog, const float* __restrict__ den,
                               const float* __restrict__ xl,
                               const int* __restrict__ src, const int* __restrict__ dst,
                               float* __restrict__ h2, int E, int Etot) {
  int wave = blockIdx.x * (blockDim.x >> 5) + (threadIdx.x >> 5);
  int l = threadIdx.x & 31;
  if (wave >= Etot) return;
  int s, d;
  if (wave < E) { s = src[wave]; d = dst[wave]; } else { s = d = wave - E; }
  float alpha = elog[wave] / den[d];
  const float4* pl = (const float4*)(xl + (size_t)s * 256 + l * 8);
  float* po = h2 + (size_t)d * 256 + l * 8;
#pragma unroll
  for (int j = 0; j < 2; ++j) {
    float4 a = pl[j];
    atomicAdd(po + 4 * j + 0, alpha * a.x);
    atomicAdd(po + 4 * j + 1, alpha * a.y);
    atomicAdd(po + 4 * j + 2, alpha * a.z);
    atomicAdd(po + 4 * j + 3, alpha * a.w);
  }
}

__global__ void k_fin2(float* __restrict__ h2, const float* __restrict__ bias, int total) {
  int i = blockIdx.x * blockDim.x + threadIdx.x;
  if (i < total) h2[i] += bias[i & 255];
}

// ---------------- classifier: wave per pair ----------------

__global__ void k_cls(const float* __restrict__ vanilla, const float* __restrict__ h2,
                      const int* __restrict__ a1, const int* __restrict__ a2,
                      const float* __restrict__ Wc, const float* __restrict__ bc,
                      float* __restrict__ out, int B) {
  int wave = blockIdx.x * (blockDim.x >> 5) + (threadIdx.x >> 5);
  int l = threadIdx.x & 31;
  if (wave >= B) return;
  int i1 = a1[wave], i2 = a2[wave];
  const float* v1 = vanilla + (size_t)i1 * 256;
  const float* v2 = vanilla + (size_t)i2 * 256;
  const float* g1 = h2 + (size_t)i1 * 256;
  const float* g2 = h2 + (size_t)i2 * 256;
  float p = 0.f;
#pragma unroll
  for (int j = l; j < 256; j += 32)
    p += v1[j] * Wc[j] + v2[j] * Wc[256 + j] + g1[j] * Wc[512 + j] + g2[j] * Wc[768 + j];
#pragma unroll
  for (int o = 16; o >= 1; o >>= 1) p += __shfl_xor(p, o, 32);
  if (l == 0) out[wave] = p + bc[0];
}

// ---------------- host launch ----------------

extern "C" void kernel_launch(void* const* d_in, const int* in_sizes, int n_in,
                              void* d_out, int out_size, void* d_ws, size_t ws_size,
                              hipStream_t stream) {
  const float* gnn_x   = (const float*)d_in[0];
  const float* vanilla = (const float*)d_in[1];
  const int*   esrc    = (const int*)d_in[2];
  const int*   edst    = (const int*)d_in[3];
  const int*   a1      = (const int*)d_in[4];
  const int*   a2      = (const int*)d_in[5];
  const float* Wl1   = (const float*)d_in[6];
  const float* bl1   = (const float*)d_in[7];
  const float* Wr1   = (const float*)d_in[8];
  const float* br1   = (const float*)d_in[9];
  const float* att1  = (const float*)d_in[10];
  const float* bias1 = (const float*)d_in[11];
  const float* Wl2   = (const float*)d_in[12];
  const float* bl2   = (const float*)d_in[13];
  const float* Wr2   = (const float*)d_in[14];
  const float* br2   = (const float*)d_in[15];
  const float* att2  = (const float*)d_in[16];
  const float* bias2 = (const float*)d_in[17];
  const float* Wc    = (const float*)d_in[18];
  const float* bc    = (const float*)d_in[19];
  float* out = (float*)d_out;

  const int IN = 256, D1 = 128, D2 = 256;
  const int N = in_sizes[0] / IN;
  const int E = in_sizes[2];
  const int B = in_sizes[4];
  const int Etot = E + N;

  char* ws = (char*)d_ws;
  size_t off = 0;
  auto alloc = [&](size_t bytes) -> void* {
    void* p = ws + off;
    off = (off + bytes + 255) & ~(size_t)255;
    return p;
  };
  _Float16* xh    = (_Float16*)alloc((size_t)N * IN * 2);
  _Float16* h1h   = (_Float16*)alloc((size_t)N * D1 * 2);
  _Float16* pWl1  = (_Float16*)alloc((size_t)IN * D1 * 2);
  _Float16* pWr1  = (_Float16*)alloc((size_t)IN * D1 * 2);
  _Float16* pWl2  = (_Float16*)alloc((size_t)D1 * D2 * 2);
  _Float16* pWr2  = (_Float16*)alloc((size_t)D1 * D2 * 2);
  float* xl1   = (float*)alloc((size_t)N * D1 * 4);
  float* xr1   = (float*)alloc((size_t)N * D1 * 4);
  float* acc1  = (float*)alloc((size_t)N * D1 * 4);
  float* xl2   = (float*)alloc((size_t)N * D2 * 4);
  float* xr2   = (float*)alloc((size_t)N * D2 * 4);
  float* h2    = (float*)alloc((size_t)N * D2 * 4);
  float* e1    = (float*)alloc((size_t)Etot * 8 * 4);
  float* emax1 = (float*)alloc((size_t)N * 8 * 4);
  float* den1  = (float*)alloc((size_t)N * 8 * 4);
  float* e2    = (float*)alloc((size_t)Etot * 4);
  float* emax2 = (float*)alloc((size_t)N * 4);
  float* den2  = (float*)alloc((size_t)N * 4);

  auto nb = [](int n, int t) { return (n + t - 1) / t; };
  const int T = 256;
  const float NINF = __builtin_huge_valf() * -1.0f;

  // ---- init accumulators / softmax state ----
  k_fill<<<nb(N * D1, T), T, 0, stream>>>(acc1, N * D1, 0.f);
  k_fill<<<nb(N * D2, T), T, 0, stream>>>(h2, N * D2, 0.f);
  k_fill<<<nb(N * 8, T), T, 0, stream>>>(emax1, N * 8, NINF);
  k_fill<<<nb(N * 8, T), T, 0, stream>>>(den1, N * 8, 0.f);
  k_fill<<<nb(N, T), T, 0, stream>>>(emax2, N, NINF);
  k_fill<<<nb(N, T), T, 0, stream>>>(den2, N, 0.f);

  // ---- f16 activations + packed WMMA weights ----
  k_f32_to_f16<<<nb(N * IN, T), T, 0, stream>>>(gnn_x, xh, N * IN);
  k_pack_w<<<nb((D1 / 16) * (IN / 32) * 32, T), T, 0, stream>>>(Wl1, pWl1, IN, D1);
  k_pack_w<<<nb((D1 / 16) * (IN / 32) * 32, T), T, 0, stream>>>(Wr1, pWr1, IN, D1);
  k_pack_w<<<nb((D2 / 16) * (D1 / 32) * 32, T), T, 0, stream>>>(Wl2, pWl2, D1, D2);
  k_pack_w<<<nb((D2 / 16) * (D1 / 32) * 32, T), T, 0, stream>>>(Wr2, pWr2, D1, D2);

  // ---- layer 1 GEMMs ----
  {
    dim3 g(nb(N, 16), D1 / 128);
    k_gemm_wmma<<<g, 256, 0, stream>>>(xh, pWl1, bl1, xl1, N, IN, D1);
    k_gemm_wmma<<<g, 256, 0, stream>>>(xh, pWr1, br1, xr1, N, IN, D1);
  }

  // ---- layer 1 attention (3 edge passes) ----
  k_att1_logits<<<nb(Etot * 8, T), T, 0, stream>>>(xl1, xr1, att1, esrc, edst, e1, emax1, E, Etot);
  k_att1_exp<<<nb(Etot * 8, T), T, 0, stream>>>(e1, emax1, den1, edst, E, Etot);
  k_att1_scatter<<<nb(Etot * 8, T), T, 0, stream>>>(e1, den1, xl1, esrc, edst, acc1, E, Etot);
  k_fin1<<<nb(N * D1, T), T, 0, stream>>>(acc1, bias1, h1h, N * D1);

  // ---- layer 2 GEMMs ----
  {
    dim3 g(nb(N, 16), D2 / 128);
    k_gemm_wmma<<<g, 256, 0, stream>>>(h1h, pWl2, bl2, xl2, N, D1, D2);
    k_gemm_wmma<<<g, 256, 0, stream>>>(h1h, pWr2, br2, xr2, N, D1, D2);
  }

  // ---- layer 2 attention ----
  k_att2_logits<<<nb(Etot, 8), 256, 0, stream>>>(xl2, xr2, att2, esrc, edst, e2, emax2, E, Etot);
  k_att2_exp<<<nb(Etot, T), T, 0, stream>>>(e2, emax2, den2, edst, E, Etot);
  k_att2_scatter<<<nb(Etot, 8), 256, 0, stream>>>(e2, den2, xl2, esrc, edst, h2, E, Etot);
  k_fin2<<<nb(N * D2, T), T, 0, stream>>>(h2, bias2, N * D2);

  // ---- classifier ----
  k_cls<<<nb(B, 8), 256, 0, stream>>>(vanilla, h2, a1, a2, Wc, bc, out, B);
}
